// LTC_82884278878575
// MI455X (gfx1250) — matrix operand order repaired and checked by reading, not compiled, not run on hardware
//
#include <hip/hip_runtime.h>

// CDNA5 / gfx1250 LTC scan.  B=128, T=2048, M=128, N=256, MOTOR=64, UNFOLDS=6.
typedef __attribute__((ext_vector_type(16))) _Float16 v16h;
typedef __attribute__((ext_vector_type(8)))  float    v8f;

#define LOG2E 1.4426950408889634f

#if __has_builtin(__builtin_amdgcn_tanhf)
#define USE_TANH 1     // confirmed: lowers to v_tanh_f32 on gfx1250
#else
#define USE_TANH 0
#endif

// ---------------------------------------------------------------------------
// Prepass: fold the sigmoid argument and weights into a float4 per synapse,
// transposed to [post][pre] so each lane loads a contiguous K run as b128.
//   tanh path:  sigmoid(s(v-m)) = 0.5 + 0.5*tanh(0.5 s (v-m))
//               p = {0.5s, -0.5s*m, 0.5w*mask, 0.5w*erev*mask}
//               elem_den = fma(p.z, tanh(fma(p.x,v,p.y)), p.z)
// ---------------------------------------------------------------------------
__global__ __launch_bounds__(256) void ltc_prepass(
    const float* __restrict__ sigma,  const float* __restrict__ mu,
    const float* __restrict__ w,      const float* __restrict__ erev,
    const float* __restrict__ mask,
    const float* __restrict__ s_sigma,const float* __restrict__ s_mu,
    const float* __restrict__ s_w,    const float* __restrict__ s_erev,
    const float* __restrict__ s_mask,
    float4* __restrict__ recP, float4* __restrict__ senP)
{
  int tid = blockIdx.x * blockDim.x + threadIdx.x;
  if (tid < 256 * 256) {
    int j = tid >> 8, i = tid & 255;
    int src = i * 256 + j;
    float ww = w[src] * mask[src];
#if USE_TANH
    float a = 0.5f * sigma[src];
    recP[j * 256 + i] = make_float4(a, -a * mu[src], 0.5f * ww,
                                    0.5f * ww * erev[src]);
#else
    float a = -sigma[src] * LOG2E;
    recP[j * 256 + i] = make_float4(a, -a * mu[src], ww, ww * erev[src]);
#endif
  } else if (tid < 256 * 256 + 128 * 256) {
    int s = tid - 65536;
    int j = s >> 7, m = s & 127;
    int src = m * 256 + j;
    float ww = s_w[src] * s_mask[src];
#if USE_TANH
    float a = 0.5f * s_sigma[src];
    senP[j * 128 + m] = make_float4(a, -a * s_mu[src], 0.5f * ww,
                                    0.5f * ww * s_erev[src]);
#else
    float a = -s_sigma[src] * LOG2E;
    senP[j * 128 + m] = make_float4(a, -a * s_mu[src], ww, ww * s_erev[src]);
#endif
  }
}

__device__ __forceinline__ void syn_eval(float4 p, float v,
                                         float& den_e, float& num_e) {
#if USE_TANH
  float th = __builtin_amdgcn_tanhf(fmaf(p.x, v, p.y));   // 1 TRANS
  den_e = fmaf(p.z, th, p.z);
  num_e = fmaf(p.w, th, p.w);
#else
  float e = __builtin_amdgcn_exp2f(fmaf(p.x, v, p.y));
  float r = __builtin_amdgcn_rcpf(1.0f + e);
  den_e = p.z * r;
  num_e = p.w * r;
#endif
}

// One K-chunk: batch-load 16 float4 params (one deep load clause, ~64 VGPRs in
// flight to hide L2 latency), evaluate synapses, accumulate via WMMA.
// rowLen = stride of the [post][pre] param matrix (256 recurrent, 128 sensory).
template <int ROWLEN>
__device__ __forceinline__ void chunk_accum(
    const float4* __restrict__ P, int j0, int k0, int lane, float vsrc,
    const v16h& ones, v8f& cden, v8f& cnum)
{
  float4 pbuf[16];
  const float4* base = P + (size_t)(j0 + (lane & 15)) * ROWLEN + k0 + 16 * (lane >> 4);
  #pragma unroll
  for (int n = 0; n < 16; ++n)
    pbuf[n] = base[n];                     // per-lane contiguous K run, global_load_b128
  v16h bden, bnum;
  #pragma unroll
  for (int n = 0; n < 16; ++n) {
    float de, nu; syn_eval(pbuf[n], __shfl(vsrc, 16 * (lane >> 4) + n), de, nu);
    bden[n] = (_Float16)de;
    bnum[n] = (_Float16)nu;
  }
  cden = __builtin_amdgcn_wmma_f32_16x16x32_f16(false, ones, false, bden,
                                                (short)0, cden, false, false);
  cnum = __builtin_amdgcn_wmma_f32_16x16x32_f16(false, ones, false, bnum,
                                                (short)0, cnum, false, false);
}

// ---------------------------------------------------------------------------
// Main scan. grid = B; block = 512 threads = 16 waves; wave w owns post-neuron
// tile [16w,16w+16). Lane L supplies B column L%16 (post j0+L%16), K rows k0+16*(L/16)+s
// (B layout: lane%16 encodes the N col, v16h element s encodes K, here k = 16*(L/16)+s).
// A = ones(f16): D accumulates per-column sums in f32 on the matrix pipe,
// co-executing with the VALU tanh chain. All D rows identical -> every lane
// reads its column sum from element 0; control flow around WMMAs is uniform.
// ---------------------------------------------------------------------------
__global__ __launch_bounds__(512) void ltc_scan(
    const float* __restrict__ x,       const float* __restrict__ h0,
    const float* __restrict__ input_w, const float* __restrict__ input_b,
    const float* __restrict__ gleak,   const float* __restrict__ vleak,
    const float* __restrict__ cm,
    const float* __restrict__ output_w,const float* __restrict__ output_b,
    const float4* __restrict__ recP,   const float4* __restrict__ senP,
    float* __restrict__ readout,       float* __restrict__ hfinal)
{
  __shared__ float v_sh[256];
  __shared__ float xi_sh[128];

  const int b     = blockIdx.x;
  const int tid   = threadIdx.x;
  const int lane  = tid & 31;
  const int wave  = tid >> 5;
  const int j0    = wave * 16;
  const int jmine = j0 + (lane & 15);

  v16h ones;
  #pragma unroll
  for (int k = 0; k < 16; ++k) ones[k] = (_Float16)1.0f;

  const float g   = gleak[jmine];
  const float gv  = g * vleak[jmine];
  const float cmt = cm[jmine] * 6.0f;   // cm * ODE_UNFOLDS

  if (tid < 256) v_sh[tid] = h0[b * 256 + tid];
  __syncthreads();

  const float* xb   = x + (size_t)b * 2048 * 128;
  float*       outb = readout + (size_t)b * 2048 * 64;

  for (int t = 0; t < 2048; ++t) {
    if (tid < 128) {
      xi_sh[tid] = xb[t * 128 + tid] * input_w[tid] + input_b[tid];
      if (t + 1 < 2048) __builtin_prefetch(&xb[(t + 1) * 128 + tid], 0, 0);
    }
    __syncthreads();

    // ---- sensory synapses: reduce over M=128 in 4 K-chunks of 32 ----
    v8f csd = {}; v8f csn = {};
    #pragma unroll 1
    for (int m0 = 0; m0 < 128; m0 += 32) {
      float xm = xi_sh[m0 + lane];
      chunk_accum<128>(senP, j0, m0, lane, xm, ones, csd, csn);
    }
    const float sden = csd[0];
    const float snum = csn[0];

    // ---- 6 semi-implicit Euler unfolds: reduce over N=256 in 8 K-chunks ----
    #pragma unroll 1
    for (int u = 0; u < 6; ++u) {
      v8f cd = {}; v8f cn = {};          // plain zero accumulators;
      #pragma unroll 1                   // sensory sums added at the end
      for (int i0 = 0; i0 < 256; i0 += 32) {
        float vi = v_sh[i0 + lane];      // conflict-free DS read
        chunk_accum<256>(recP, j0, i0, lane, vi, ones, cd, cn);
      }
      float vj   = v_sh[jmine];
      float num  = cn[0] + snum;
      float den  = cd[0] + sden;
      float vnew = (cmt * vj + gv + num) *
                   __builtin_amdgcn_rcpf(cmt + g + den + 1e-8f);
      __syncthreads();                   // all reads of old v complete
      if (lane < 16) v_sh[jmine] = vnew;
      __syncthreads();                   // new v visible to all waves
    }

    if (tid < 64) outb[t * 64 + tid] = v_sh[tid] * output_w[tid] + output_b[tid];
    __syncthreads();
  }

  if (tid < 256) hfinal[b * 256 + tid] = v_sh[tid];
}

// ---------------------------------------------------------------------------
extern "C" void kernel_launch(void* const* d_in, const int* in_sizes, int n_in,
                              void* d_out, int out_size, void* d_ws, size_t ws_size,
                              hipStream_t stream) {
  const float* x        = (const float*)d_in[0];
  const float* h0       = (const float*)d_in[1];
  const float* input_w  = (const float*)d_in[2];
  const float* input_b  = (const float*)d_in[3];
  const float* gleak    = (const float*)d_in[4];
  const float* vleak    = (const float*)d_in[5];
  const float* cm       = (const float*)d_in[6];
  const float* sigma    = (const float*)d_in[7];
  const float* mu       = (const float*)d_in[8];
  const float* w        = (const float*)d_in[9];
  const float* erev     = (const float*)d_in[10];
  const float* mask     = (const float*)d_in[11];
  const float* s_sigma  = (const float*)d_in[12];
  const float* s_mu     = (const float*)d_in[13];
  const float* s_w      = (const float*)d_in[14];
  const float* s_erev   = (const float*)d_in[15];
  const float* s_mask   = (const float*)d_in[16];
  const float* output_w = (const float*)d_in[17];
  const float* output_b = (const float*)d_in[18];

  float4* recP = (float4*)d_ws;            // 65536 float4 = 1.0 MB
  float4* senP = recP + 256 * 256;         // 32768 float4 = 0.5 MB

  float* readout = (float*)d_out;
  float* hfinal  = readout + (size_t)128 * 2048 * 64;

  ltc_prepass<<<dim3(384), dim3(256), 0, stream>>>(
      sigma, mu, w, erev, mask, s_sigma, s_mu, s_w, s_erev, s_mask, recP, senP);

  ltc_scan<<<dim3(128), dim3(512), 0, stream>>>(
      x, h0, input_w, input_b, gleak, vleak, cm,
      output_w, output_b, recP, senP, readout, hfinal);
}
